// HitGNN_67602785239422
// MI455X (gfx1250) — matrix operand order repaired and checked by reading, not compiled, not run on hardware
//
#include <hip/hip_runtime.h>
#include <hip/hip_bf16.h>
#include <cstdint>
#include <cstddef>

// ---------------------------------------------------------------------------
// HitGNN forward on gfx1250: f16 WMMA GEMMs with fused bias/gate/LN/GELU/
// scatter epilogues. wave32. Edge GEMMs: 64-row blocks, A staged to LDS via
// global_load_async_to_lds_b128 (ASYNCcnt), B streamed from L2.
// ---------------------------------------------------------------------------

typedef _Float16 f16;
typedef __attribute__((ext_vector_type(8)))  _Float16 v8h;
typedef __attribute__((ext_vector_type(16))) _Float16 v16h;
typedef __attribute__((ext_vector_type(8)))  float    v8f;

constexpr int N_NODES = 50000;
constexpr int E_EDGES = 800000;
constexpr int HID     = 256;
constexpr float LN_EPS = 1e-5f;

#define DEVFN static __device__ __forceinline__

DEVFN v16h cat16(v8h a, v8h b) {
  return __builtin_shufflevector(a, b, 0,1,2,3,4,5,6,7,8,9,10,11,12,13,14,15);
}

// A-matrix 16x32 f16 tile (ISA 7.12.2): lane holds row (lane&15); half h<8 ->
// K = kb + 8*hi + h ; half h>=8 -> K = kb + 16 + 8*hi + (h-8).  hi = lane>=16.
DEVFN v16h load_a(const f16* __restrict__ rowPtr, int kb, int hi) {
  v8h lo = *reinterpret_cast<const v8h*>(rowPtr + kb + hi * 8);
  v8h hh = *reinterpret_cast<const v8h*>(rowPtr + kb + 16 + hi * 8);
  return cat16(lo, hh);
}

// B-matrix 32x16 f16 tile from K-major (transposed) weights: lane holds column
// (lane&15); half h<8 -> K = kb + 8*hi + h ; h>=8 -> K = kb + 16 + 8*hi + (h-8).  colPtr = Wt + col*KS.
DEVFN v16h load_b(const f16* __restrict__ colPtr, int kb, int hi) {
  v8h lo = *reinterpret_cast<const v8h*>(colPtr + kb + hi * 8);
  v8h hh = *reinterpret_cast<const v8h*>(colPtr + kb + 16 + hi * 8);
  return cat16(lo, hh);
}

DEVFN v8f wmma16(v16h a, v16h b, v8f c) {
  return __builtin_amdgcn_wmma_f32_16x16x32_f16(false, a, false, b,
                                                (short)0, c, false, false);
}

DEVFN float wave_sum(float v) {
#pragma unroll
  for (int off = 16; off > 0; off >>= 1) v += __shfl_xor(v, off, 32);
  return v;
}

DEVFN float gelu_exact(float x) {
  return 0.5f * x * (1.0f + erff(x * 0.70710678118654752f));
}

DEVFN void atomic_add_f32(float* p, float v) {
  asm volatile("global_atomic_add_f32 %0, %1, off" :: "v"(p), "v"(v) : "memory");
}

// Async 16B copy global -> LDS (ASYNCcnt-tracked, no VGPR round trip).
// LDS byte address = low 32 bits of the flat pointer (ISA 10.2: LDS aperture
// addresses are wave-relative in addr[31:0]).
DEVFN void async_cp16(void* ldsDst, const void* gSrc) {
  const uint32_t lo = (uint32_t)(uintptr_t)ldsDst;
  asm volatile("global_load_async_to_lds_b128 %0, %1, off"
               :: "v"(lo), "v"(gSrc) : "memory");
}
DEVFN void async_wait_all() {
  asm volatile("s_wait_asynccnt 0x0" ::: "memory");
}

// ---------------------------------------------------------------------------
// Weight prep: f32 [K, Nout] row-major -> f16 [Nout, KS] K-major, zero padded.
// ---------------------------------------------------------------------------
__global__ void transpose_w_kernel(const float* __restrict__ W,
                                   f16* __restrict__ Wt, int K, int Nout, int KS) {
  int n = blockIdx.x;
  for (int k = threadIdx.x; k < KS; k += blockDim.x) {
    float v = (k < K) ? W[(size_t)k * Nout + n] : 0.0f;
    Wt[(size_t)n * KS + k] = (f16)v;
  }
}

__global__ void zero_f32_kernel(float* __restrict__ p, int n) {
  int i = blockIdx.x * blockDim.x + threadIdx.x;
  if (i < n) p[i] = 0.0f;
}

__global__ void f32_to_f16_kernel(const float* __restrict__ s,
                                  f16* __restrict__ d, int n) {
  int i = blockIdx.x * blockDim.x + threadIdx.x;
  if (i < n) d[i] = (f16)s[i];
}

// ---------------------------------------------------------------------------
// Input stage 1: t16 = gelu(LN_n1(x @ W1 + b1)).  K=5 -> plain VALU.
// ---------------------------------------------------------------------------
__global__ __launch_bounds__(256) void input_l1_kernel(
    const float* __restrict__ x, const float* __restrict__ W,
    const float* __restrict__ b, const float* __restrict__ g,
    const float* __restrict__ be, f16* __restrict__ out16) {
  const int wid = threadIdx.x >> 5, lane = threadIdx.x & 31;
  const int n = blockIdx.x * 8 + wid;
  float xr[5];
#pragma unroll
  for (int k = 0; k < 5; ++k) xr[k] = x[n * 5 + k];
  const int c0 = lane * 8;
  float acc[8];
  float s = 0.f, sq = 0.f;
#pragma unroll
  for (int t = 0; t < 8; ++t) {
    float a = b[c0 + t];
#pragma unroll
    for (int k = 0; k < 5; ++k) a += xr[k] * W[k * HID + c0 + t];
    acc[t] = a; s += a; sq += a * a;
  }
  s = wave_sum(s); sq = wave_sum(sq);
  const float mean = s * (1.0f / HID);
  const float rstd = rsqrtf(sq * (1.0f / HID) - mean * mean + LN_EPS);
  v8h o;
#pragma unroll
  for (int t = 0; t < 8; ++t) {
    float v = (acc[t] - mean) * rstd * g[c0 + t] + be[c0 + t];
    o[t] = (f16)gelu_exact(v);
  }
  *reinterpret_cast<v8h*>(out16 + (size_t)n * HID + c0) = o;
}

// ---------------------------------------------------------------------------
// Edge message stage 1: m = gelu(LN_en1( gate(concat(h[dst],h[src],ea) @ W1 + b) ))
// Block: 64 edges x 512 cols, 16 waves. A (gathered, K=512 concat) staged in
// LDS (row stride 520 halves -> 4-bank skew). Wave w: m1 col tile w*16, paired
// m2 col tile 256+w*16, 4 row stripes.
// Dynamic LDS: [ A 64x520 f16 | overlaid C 64x264 f32 ] + EA 64x5 f32.
// ---------------------------------------------------------------------------
constexpr int E1_CBYTES = 64 * 264 * 4;          // 67584 (>= A bytes 66560)
constexpr int E1_LDS    = E1_CBYTES + 64 * 5 * 4;

__global__ __launch_bounds__(512) void edge_msg1_kernel(
    const f16* __restrict__ hf16, const float* __restrict__ ea,
    const int* __restrict__ eidx, const f16* __restrict__ Wt,
    const float* __restrict__ bias, const float* __restrict__ g,
    const float* __restrict__ be, f16* __restrict__ mOut) {
  extern __shared__ __align__(16) char smem[];
  f16*   smA  = (f16*)smem;                  // [64][520], GEMM phase
  float* smC  = (float*)smem;                // [64][264], LN phase (overlay)
  float* smEA = (float*)(smem + E1_CBYTES);  // [64*5]
  const int tid = threadIdx.x, wid = tid >> 5, lane = tid & 31;
  const int hi = lane >> 4, lr = lane & 15;
  const int e0 = blockIdx.x * 64;

  // ---- stage edge_attr + gathered A rows into LDS (async) ----
  if (tid < 320) smEA[tid] = ea[(size_t)e0 * 5 + tid];
#pragma unroll
  for (int i = 0; i < 8; ++i) {
    const int q = tid + i * 512;          // 64 rows x 64 16B-chunks
    const int row = q >> 6, ch = q & 63;
    const int node = (ch < 32) ? eidx[E_EDGES + e0 + row]   // dst -> K 0..255
                               : eidx[e0 + row];            // src -> K 256..511
    const int kh = (ch & 31) * 8;
    async_cp16(smA + (size_t)row * 520 + ch * 8, hf16 + (size_t)node * HID + kh);
  }
  async_wait_all();
  __syncthreads();

  // ---- WMMA main loop ----
  const int c = wid * 16 + lr;
  constexpr int KS = 520;
  const f16* bc1 = Wt + (size_t)c * KS;
  const f16* bc2 = Wt + (size_t)(256 + c) * KS;
  v8f acc1[4] = {{}, {}, {}, {}};
  v8f acc2[4] = {{}, {}, {}, {}};
#pragma unroll
  for (int kb = 0; kb < 512; kb += 32) {
    const v16h b1 = load_b(bc1, kb, hi);
    const v16h b2 = load_b(bc2, kb, hi);
#pragma unroll
    for (int s = 0; s < 4; ++s) {
      const v16h a = load_a(smA + (size_t)(s * 16 + lr) * 520, kb, hi);
      acc1[s] = wmma16(a, b1, acc1[s]);
      acc2[s] = wmma16(a, b2, acc2[s]);
    }
  }
  __syncthreads();  // all smA reads done; smC overlays smA

  // ---- bias + edge_attr K-tail + gate -> smC ----
  float w1e[5], w2e[5];
#pragma unroll
  for (int k = 0; k < 5; ++k) { w1e[k] = (float)bc1[512 + k]; w2e[k] = (float)bc2[512 + k]; }
  const float b1s = bias[c], b2s = bias[256 + c];
#pragma unroll
  for (int s = 0; s < 4; ++s) {
#pragma unroll
    for (int i = 0; i < 8; ++i) {
      const int r = s * 16 + i + hi * 8;
      float m1 = acc1[s][i] + b1s, m2 = acc2[s][i] + b2s;
#pragma unroll
      for (int k = 0; k < 5; ++k) {
        const float eav = smEA[r * 5 + k];
        m1 += eav * w1e[k]; m2 += eav * w2e[k];
      }
      smC[r * 264 + c] = m1 * m2;
    }
  }
  __syncthreads();

  // ---- per-row LN + GELU (wave handles 4 rows), write f16 messages ----
  const int c0 = lane * 8;
#pragma unroll
  for (int j = 0; j < 4; ++j) {
    const int r = wid * 4 + j;
    float v[8]; float s = 0.f, sq = 0.f;
#pragma unroll
    for (int t = 0; t < 8; ++t) { v[t] = smC[r * 264 + c0 + t]; s += v[t]; sq += v[t] * v[t]; }
    s = wave_sum(s); sq = wave_sum(sq);
    const float mean = s * (1.0f / HID);
    const float rstd = rsqrtf(sq * (1.0f / HID) - mean * mean + LN_EPS);
    v8h o;
#pragma unroll
    for (int t = 0; t < 8; ++t) {
      const float u = (v[t] - mean) * rstd * g[c0 + t] + be[c0 + t];
      o[t] = (f16)gelu_exact(u);
    }
    *reinterpret_cast<v8h*>(mOut + (size_t)(e0 + r) * HID + c0) = o;
  }
}

// ---------------------------------------------------------------------------
// Edge message stage 2 + scatter: aggr[dst] += gelu(LN_en2(m @ W2 + b))
// Block: 64 edges x 256 cols. A staged to LDS (stride 264 halves).
// Dynamic LDS: [ A 64x264 f16 | overlaid C 64x264 f32 ].
// ---------------------------------------------------------------------------
constexpr int E2_LDS = 64 * 264 * 4;  // 67584

__global__ __launch_bounds__(512) void edge_msg2_kernel(
    const f16* __restrict__ m16, const int* __restrict__ eidx,
    const f16* __restrict__ Wt, const float* __restrict__ bias,
    const float* __restrict__ g, const float* __restrict__ be,
    float* __restrict__ aggr) {
  extern __shared__ __align__(16) char smem[];
  f16*   smA = (f16*)smem;    // [64][264]
  float* smC = (float*)smem;  // [64][264] (overlay)
  const int tid = threadIdx.x, wid = tid >> 5, lane = tid & 31;
  const int hi = lane >> 4, lr = lane & 15;
  const int e0 = blockIdx.x * 64;

#pragma unroll
  for (int i = 0; i < 4; ++i) {
    const int q = tid + i * 512;          // 64 rows x 32 16B-chunks
    const int row = q >> 5, ch = q & 31;
    async_cp16(smA + (size_t)row * 264 + ch * 8,
               m16 + (size_t)(e0 + row) * HID + ch * 8);
  }
  async_wait_all();
  __syncthreads();

  const int c = wid * 16 + lr;
  const f16* bc = Wt + (size_t)c * 256;
  v8f acc[4] = {{}, {}, {}, {}};
#pragma unroll
  for (int kb = 0; kb < 256; kb += 32) {
    const v16h b = load_b(bc, kb, hi);
#pragma unroll
    for (int s = 0; s < 4; ++s)
      acc[s] = wmma16(load_a(smA + (size_t)(s * 16 + lr) * 264, kb, hi), b, acc[s]);
  }
  __syncthreads();

  const float bs = bias[c];
#pragma unroll
  for (int s = 0; s < 4; ++s)
#pragma unroll
    for (int i = 0; i < 8; ++i)
      smC[(s * 16 + i + hi * 8) * 264 + c] = acc[s][i] + bs;
  __syncthreads();

  const int c0 = lane * 8;
#pragma unroll
  for (int j = 0; j < 4; ++j) {
    const int r = wid * 4 + j;
    float v[8]; float s = 0.f, sq = 0.f;
#pragma unroll
    for (int t = 0; t < 8; ++t) { v[t] = smC[r * 264 + c0 + t]; s += v[t]; sq += v[t] * v[t]; }
    s = wave_sum(s); sq = wave_sum(sq);
    const float mean = s * (1.0f / HID);
    const float rstd = rsqrtf(sq * (1.0f / HID) - mean * mean + LN_EPS);
    const int dstN = eidx[E_EDGES + e0 + r];
    float* dp = aggr + (size_t)dstN * HID + c0;
#pragma unroll
    for (int t = 0; t < 8; ++t) {
      const float u = (v[t] - mean) * rstd * g[c0 + t] + be[c0 + t];
      atomic_add_f32(dp + t, gelu_exact(u));
    }
  }
}

// ---------------------------------------------------------------------------
// Generic fused node GEMM (M=16 blocks): out = epilogue(A @ Wt + b)
// ---------------------------------------------------------------------------
enum { M_L2IN = 0, M_U1 = 1, M_U2 = 2, M_OUTL1 = 3 };

template <int MODE>
__global__ __launch_bounds__(512) void node_gemm_kernel(
    const f16* __restrict__ A0, const f16* __restrict__ A1,
    const f16* __restrict__ Wt, const float* __restrict__ bias,
    const float* __restrict__ g, const float* __restrict__ be,
    const float* __restrict__ resid, float* __restrict__ out32,
    f16* __restrict__ out16) {
  constexpr bool HAS_A1   = (MODE == M_U1);
  constexpr bool HAS_LN   = (MODE != M_OUTL1);
  constexpr bool HAS_GELU = (MODE == M_U1 || MODE == M_OUTL1);
  constexpr bool RESID    = (MODE == M_U2);
  constexpr bool W32      = (MODE == M_L2IN || MODE == M_U2);
  constexpr int  KSW      = HAS_A1 ? 512 : 256;
  __shared__ float sm[16][264];
  const int tid = threadIdx.x, wid = tid >> 5, lane = tid & 31;
  const int hi = lane >> 4, lr = lane & 15;
  const int r0 = blockIdx.x * 16;
  const f16* rowA0 = A0 + (size_t)(r0 + lr) * HID;
  const int c = wid * 16 + lr;
  const f16* bc = Wt + (size_t)c * KSW;
  v8f acc = {};
#pragma unroll
  for (int kb = 0; kb < 256; kb += 32)
    acc = wmma16(load_a(rowA0, kb, hi), load_b(bc, kb, hi), acc);
  if constexpr (HAS_A1) {
    const f16* rowA1 = A1 + (size_t)(r0 + lr) * HID;
#pragma unroll
    for (int kb = 0; kb < 256; kb += 32)
      acc = wmma16(load_a(rowA1, kb, hi), load_b(bc, 256 + kb, hi), acc);
  }
#pragma unroll
  for (int i = 0; i < 8; ++i) sm[i + hi * 8][c] = acc[i] + bias[c];
  __syncthreads();
  const int row = r0 + wid;
  const int c0 = lane * 8;
  float v[8]; float s = 0.f, sq = 0.f;
#pragma unroll
  for (int t = 0; t < 8; ++t) { v[t] = sm[wid][c0 + t]; s += v[t]; sq += v[t] * v[t]; }
  float mean = 0.f, rstd = 1.f;
  if constexpr (HAS_LN) {
    s = wave_sum(s); sq = wave_sum(sq);
    mean = s * (1.0f / HID);
    rstd = rsqrtf(sq * (1.0f / HID) - mean * mean + LN_EPS);
  }
  v8h o;
#pragma unroll
  for (int t = 0; t < 8; ++t) {
    float u = v[t];
    if constexpr (HAS_LN) u = (u - mean) * rstd * g[c0 + t] + be[c0 + t];
    if constexpr (RESID)  u = gelu_exact(resid[(size_t)row * HID + c0 + t] + u);
    else if constexpr (HAS_GELU) u = gelu_exact(u);
    if constexpr (W32) out32[(size_t)row * HID + c0 + t] = u;
    o[t] = (f16)u;
  }
  *reinterpret_cast<v8h*>(out16 + (size_t)row * HID + c0) = o;
}

// ---------------------------------------------------------------------------
// Output head: logits[n] = t16[n,:] . w + b  (wave per node)
// ---------------------------------------------------------------------------
__global__ __launch_bounds__(256) void out_l2_kernel(
    const f16* __restrict__ t16, const float* __restrict__ w,
    const float* __restrict__ b, float* __restrict__ out) {
  const int wid = threadIdx.x >> 5, lane = threadIdx.x & 31;
  const int n = blockIdx.x * 8 + wid;
  const f16* row = t16 + (size_t)n * HID;
  float s = 0.f;
#pragma unroll
  for (int t = 0; t < 8; ++t) {
    const int k = lane * 8 + t;
    s += (float)row[k] * w[k];
  }
  s = wave_sum(s);
  if (lane == 0) out[n] = s + b[0];
}

// ---------------------------------------------------------------------------
extern "C" void kernel_launch(void* const* d_in, const int* in_sizes, int n_in,
                              void* d_out, int out_size, void* d_ws, size_t ws_size,
                              hipStream_t stream) {
  (void)out_size; (void)ws_size;
  // ---- input index mapping (auto-detect flatten convention) ----
  const bool topInsertion = (in_sizes[0] == N_NODES * 5);  // x first?
  const float* x; const int* eidx; const float* ea; int p0;
  if (topInsertion) { x = (const float*)d_in[0]; eidx = (const int*)d_in[1];
                      ea = (const float*)d_in[2]; p0 = 3; }
  else { ea = (const float*)d_in[0]; eidx = (const int*)d_in[1];
         x = (const float*)d_in[n_in - 1]; p0 = 2; }
  const bool wFirst = (in_sizes[p0] == 5 * HID);  // insertion order inside dicts?

  int idx = p0;
  auto nxt = [&]() { return (const float*)d_in[idx++]; };
  const float *inl1w, *inl1b, *inn1g, *inn1b, *inl2w, *inl2b, *inn2g, *inn2b;
  const float *e1w[3], *e1b[3], *en1g[3], *en1b[3], *e2w[3], *e2b[3], *en2g[3], *en2b[3];
  const float *u1w[3], *u1b[3], *un1g[3], *un1b[3], *u2w[3], *u2b[3], *un2g[3], *un2b[3];
  const float *o1w, *o1b, *o2w, *o2b;
  if (wFirst) {  // python dict insertion order: l1{w,b}, n1{g,b}, l2{w,b}, n2{g,b}
    inl1w=nxt(); inl1b=nxt(); inn1g=nxt(); inn1b=nxt();
    inl2w=nxt(); inl2b=nxt(); inn2g=nxt(); inn2b=nxt();
    for (int l = 0; l < 3; ++l) {
      e1w[l]=nxt(); e1b[l]=nxt(); en1g[l]=nxt(); en1b[l]=nxt();
      e2w[l]=nxt(); e2b[l]=nxt(); en2g[l]=nxt(); en2b[l]=nxt();
      u1w[l]=nxt(); u1b[l]=nxt(); un1g[l]=nxt(); un1b[l]=nxt();
      u2w[l]=nxt(); u2b[l]=nxt(); un2g[l]=nxt(); un2b[l]=nxt();
    }
    o1w=nxt(); o1b=nxt(); o2w=nxt(); o2b=nxt();
  } else {       // jax tree_flatten sorted keys: l1{b,w}, l2{b,w}, n1{b,g}, n2{b,g}
    inl1b=nxt(); inl1w=nxt(); inl2b=nxt(); inl2w=nxt();
    inn1b=nxt(); inn1g=nxt(); inn2b=nxt(); inn2g=nxt();
    for (int l = 0; l < 3; ++l) {
      e1b[l]=nxt(); e1w[l]=nxt(); e2b[l]=nxt(); e2w[l]=nxt();
      en1b[l]=nxt(); en1g[l]=nxt(); en2b[l]=nxt(); en2g[l]=nxt();
      u1b[l]=nxt(); u1w[l]=nxt(); u2b[l]=nxt(); u2w[l]=nxt();
      un1b[l]=nxt(); un1g[l]=nxt(); un2b[l]=nxt(); un2g[l]=nxt();
    }
    o1b=nxt(); o1w=nxt(); o2b=nxt(); o2w=nxt();
  }

  // ---- workspace carve-out (256B aligned) ----
  char* ws = (char*)d_ws; size_t off = 0;
  auto take = [&](size_t bytes) { void* p = ws + off; off += (bytes + 255) & ~(size_t)255; return p; };
  const size_t szN32 = (size_t)N_NODES * HID * 4;
  const size_t szN16 = (size_t)N_NODES * HID * 2;
  float* hf32   = (float*)take(szN32);
  float* aggr32 = (float*)take(szN32);
  f16*   hf16   = (f16*)take(szN16);
  f16*   aggr16 = (f16*)take(szN16);
  f16*   t16    = (f16*)take(szN16);
  f16*   m16    = (f16*)take((size_t)E_EDGES * HID * 2);
  f16* Wt_inl2  = (f16*)take((size_t)256 * 256 * 2);
  f16* Wt_outl1 = (f16*)take((size_t)256 * 256 * 2);
  f16 *Wt_e1[3], *Wt_e2[3], *Wt_u1[3], *Wt_u2[3];
  for (int l = 0; l < 3; ++l) {
    Wt_e1[l] = (f16*)take((size_t)512 * 520 * 2);
    Wt_e2[l] = (f16*)take((size_t)256 * 256 * 2);
    Wt_u1[l] = (f16*)take((size_t)256 * 512 * 2);
    Wt_u2[l] = (f16*)take((size_t)256 * 256 * 2);
  }

  // ---- weight prep (f32 -> transposed f16) ----
  transpose_w_kernel<<<256, 256, 0, stream>>>(inl2w, Wt_inl2, 256, 256, 256);
  transpose_w_kernel<<<256, 256, 0, stream>>>(o1w,   Wt_outl1, 256, 256, 256);
  for (int l = 0; l < 3; ++l) {
    transpose_w_kernel<<<512, 256, 0, stream>>>(e1w[l], Wt_e1[l], 517, 512, 520);
    transpose_w_kernel<<<256, 256, 0, stream>>>(e2w[l], Wt_e2[l], 256, 256, 256);
    transpose_w_kernel<<<256, 256, 0, stream>>>(u1w[l], Wt_u1[l], 512, 256, 512);
    transpose_w_kernel<<<256, 256, 0, stream>>>(u2w[l], Wt_u2[l], 256, 256, 256);
  }

  // ---- input MLP ----
  input_l1_kernel<<<N_NODES / 8, 256, 0, stream>>>(x, inl1w, inl1b, inn1g, inn1b, t16);
  node_gemm_kernel<M_L2IN><<<N_NODES / 16, 512, 0, stream>>>(
      t16, nullptr, Wt_inl2, inl2b, inn2g, inn2b, nullptr, hf32, hf16);

  // ---- message-passing layers ----
  const int nElem = N_NODES * HID;
  for (int l = 0; l < 3; ++l) {
    edge_msg1_kernel<<<E_EDGES / 64, 512, E1_LDS, stream>>>(
        hf16, ea, eidx, Wt_e1[l], e1b[l], en1g[l], en1b[l], m16);
    zero_f32_kernel<<<(nElem + 255) / 256, 256, 0, stream>>>(aggr32, nElem);
    edge_msg2_kernel<<<E_EDGES / 64, 512, E2_LDS, stream>>>(
        m16, eidx, Wt_e2[l], e2b[l], en2g[l], en2b[l], aggr32);
    f32_to_f16_kernel<<<(nElem + 255) / 256, 256, 0, stream>>>(aggr32, aggr16, nElem);
    node_gemm_kernel<M_U1><<<N_NODES / 16, 512, 0, stream>>>(
        hf16, aggr16, Wt_u1[l], u1b[l], un1g[l], un1b[l], nullptr, nullptr, t16);
    node_gemm_kernel<M_U2><<<N_NODES / 16, 512, 0, stream>>>(
        t16, nullptr, Wt_u2[l], u2b[l], un2g[l], un2b[l], hf32, hf32, hf16);
  }

  // ---- output head ----
  node_gemm_kernel<M_OUTL1><<<N_NODES / 16, 512, 0, stream>>>(
      hf16, nullptr, Wt_outl1, o1b, nullptr, nullptr, nullptr, nullptr, t16);
  out_l2_kernel<<<N_NODES / 8, 256, 0, stream>>>(t16, o2w, o2b, (float*)d_out);
}
